// HTATeacher_72911364817310
// MI455X (gfx1250) — matrix operand order repaired and checked by reading, not compiled, not run on hardware
//
#include <hip/hip_runtime.h>
#include <math.h>

// =====================================================================
// HTATeacher forward for MI455X (gfx1250, wave32, WMMA).
// All large GEMMs: bf16 x bf16 -> f32 via v_wmma_f32_16x16x32_bf16,
// 2x4 16x16 tiles per wave (32x64), 8 waves/block (64x256 block tile),
// uniform fast path (no per-lane divergence in the K loop), bf16 outputs
// fused into GEMM / softmax / LN / combine to avoid extra HBM passes.
//
// Input order (jax pytree flatten, dict keys sorted):
//   0:x  1:H  2:U_r  3:cls_b 4:cls_w 5:in_b 6:in_w
//   then per layer l (base = 7 + 14*l):
//   +0 Wge +1 Wgn +2 Wgs +3 Wk +4 Wout +5 Wq +6 Wspec +7 Wv
//   +8 bge +9 bgn +10 bgs +11 bout +12 ln_b +13 ln_g
// Output: logits (3072*7) | H_dropped (3072*1536) | drop_mask (1536), f32.
// =====================================================================

typedef __attribute__((ext_vector_type(16))) __bf16 v16bf;
typedef __attribute__((ext_vector_type(8)))  float  v8f;
typedef __attribute__((ext_vector_type(8)))  unsigned short u16x8;

union FragBF { v16bf v; unsigned short u[16]; };

__device__ __forceinline__ unsigned short f2bf(float f) {
  unsigned u = __float_as_uint(f);
  unsigned r = u + 0x7FFFu + ((u >> 16) & 1u);      // round-to-nearest-even
  return (unsigned short)(r >> 16);
}

__device__ __forceinline__ float sigmoidf_(float z) {
  return 1.0f / (1.0f + __expf(-z));
}

__device__ __forceinline__ v8f wmma_bf16(const FragBF& a, const FragBF& b, v8f c) {
  return __builtin_amdgcn_wmma_f32_16x16x32_bf16(false, a.v, false, b.v,
                                                 (short)0, c, false, false);
}

__device__ __forceinline__ FragBF mk_frag(u16x8 lo, u16x8 hi) {
  FragBF f;
#pragma unroll
  for (int i = 0; i < 8; ++i) { f.u[i] = lo[i]; f.u[8 + i] = hi[i]; }
  return f;
}

// ---------------- checked WMMA fragment loaders (contiguous k = 16*(lane/16) + i) ----------------
__device__ __forceinline__ FragBF load_frag_a(const unsigned short* __restrict__ A, int lda,
                                              int row0, int k0, int Rmax, int Kmax) {
  const int lane = threadIdx.x & 31;
  FragBF f;
#pragma unroll
  for (int i = 0; i < 16; ++i) f.u[i] = 0;
  const int r  = row0 + (lane & 15);
  const int kb = k0 + ((lane >> 4) << 4);
  if (r < Rmax) {
    const unsigned short* p = A + (size_t)r * lda;
#pragma unroll
    for (int i = 0; i < 8; ++i) {
      if (kb + i      < Kmax) f.u[i]     = p[kb + i];
      if (kb + 8 + i < Kmax) f.u[8 + i] = p[kb + 8 + i];
    }
  }
  return f;
}

__device__ __forceinline__ FragBF load_frag_b_nt(const unsigned short* __restrict__ B, int ldb,
                                                 int col0, int k0, int Cmax, int Kmax) {
  const int lane = threadIdx.x & 31;
  FragBF f;
#pragma unroll
  for (int i = 0; i < 16; ++i) f.u[i] = 0;
  const int n  = col0 + (lane & 15);
  const int kb = k0 + ((lane >> 4) << 4);
  if (n < Cmax) {
    const unsigned short* p = B + (size_t)n * ldb + kb;
#pragma unroll
    for (int i = 0; i < 16; ++i)
      if (kb + i < Kmax) f.u[i] = p[i];
  }
  return f;
}

__device__ __forceinline__ FragBF load_frag_b_nn(const unsigned short* __restrict__ B, int ldb,
                                                 int col0, int k0, int Cmax, int Kmax) {
  const int lane = threadIdx.x & 31;
  FragBF f;
#pragma unroll
  for (int i = 0; i < 16; ++i) f.u[i] = 0;
  const int n  = col0 + (lane & 15);
  const int kb = k0 + ((lane >> 4) << 4);
  if (n < Cmax) {
#pragma unroll 4
    for (int i = 0; i < 16; ++i) {
      const int k = kb + i;
      if (k < Kmax) f.u[i] = B[(size_t)k * ldb + n];
    }
  }
  return f;
}

// ---------------- C tile store (optionally mirrors bf16) ----------------
__device__ __forceinline__ void store_tile(float* __restrict__ C,
                                           unsigned short* __restrict__ Cb, int ldc,
                                           int row0, int col0, int Mdim, int Ndim,
                                           const v8f& acc, const float* __restrict__ bias,
                                           bool chk) {
  const int lane  = threadIdx.x & 31;
  const int rbase = row0 + ((lane >> 4) << 3);
  const int cc    = col0 + (lane & 15);
  if (!chk || cc < Ndim) {
    const float bv = bias ? bias[cc] : 0.0f;
#pragma unroll
    for (int j = 0; j < 8; ++j) {
      const int r = rbase + j;
      if (!chk || r < Mdim) {
        const float v = acc[j] + bv;
        C[(size_t)r * ldc + cc] = v;
        if (Cb) Cb[(size_t)r * ldc + cc] = f2bf(v);
      }
    }
  }
}

// ---------------- generic bf16 GEMM, f32 accumulate ----------------
// C(M x N) = A(M x K) * op(B) + bias; BT_ROWMAJOR: B stored (N x K) (A*B^T).
// 8 waves (2 along M x 4 along N); each wave computes a 32x64 patch = 2x4 WMMA tiles.
template <bool BT_ROWMAJOR>
__global__ void __launch_bounds__(256)
gemm_bf16_ker(const unsigned short* __restrict__ A, int lda,
              const unsigned short* __restrict__ B, int ldb,
              float* __restrict__ C, int ldc,
              unsigned short* __restrict__ Cb,
              int Mdim, int Ndim, int Kdim,
              const float* __restrict__ bias) {
  const int lane = threadIdx.x & 31;
  const int wave = threadIdx.x >> 5;                  // 0..7 -> 2 x 4
  const int row0 = blockIdx.y * 64  + (wave >> 2) * 32;
  const int col0 = blockIdx.x * 256 + (wave & 3) * 64;
  if (row0 >= Mdim || col0 >= Ndim) return;           // uniform per wave
  const v8f vzero = {};
  v8f acc[2][4];
#pragma unroll
  for (int i = 0; i < 2; ++i)
#pragma unroll
    for (int j = 0; j < 4; ++j) acc[i][j] = vzero;
  const bool full = (row0 + 32 <= Mdim) && (col0 + 64 <= Ndim);
  int k0 = 0;
  if (full) {
    // Uniform fast path: unguarded vector loads, zero per-lane branching.
    const int kbA = (lane >> 4) << 4;
    const int kbB = (lane >> 4) << 4;
    const unsigned short* pA0 = A + (size_t)(row0 + (lane & 15)) * lda + kbA;
    const unsigned short* pA1 = pA0 + (size_t)16 * lda;
    const unsigned short* pB[4];
#pragma unroll
    for (int j = 0; j < 4; ++j) {
      if (BT_ROWMAJOR) pB[j] = B + (size_t)(col0 + j * 16 + (lane & 15)) * ldb + kbB;
      else             pB[j] = B + (col0 + j * 16 + (lane & 15));
    }
#pragma unroll 2
    for (; k0 + 32 <= Kdim; k0 += 32) {
      if (k0 + 32 < Kdim) {                           // global_prefetch_b8 next K slab
        __builtin_prefetch(pA0 + k0 + 32, 0, 1);
        __builtin_prefetch(pA1 + k0 + 32, 0, 1);
        if (BT_ROWMAJOR) {
#pragma unroll
          for (int j = 0; j < 4; ++j) __builtin_prefetch(pB[j] + k0 + 32, 0, 1);
        }
      }
      FragBF a0 = mk_frag(*(const u16x8*)(pA0 + k0), *(const u16x8*)(pA0 + k0 + 8));
      FragBF a1 = mk_frag(*(const u16x8*)(pA1 + k0), *(const u16x8*)(pA1 + k0 + 8));
      FragBF b[4];
      if (BT_ROWMAJOR) {
#pragma unroll
        for (int j = 0; j < 4; ++j)
          b[j] = mk_frag(*(const u16x8*)(pB[j] + k0), *(const u16x8*)(pB[j] + k0 + 8));
      } else {
#pragma unroll
        for (int j = 0; j < 4; ++j) {
#pragma unroll
          for (int i = 0; i < 16; ++i)
            b[j].u[i] = pB[j][(size_t)(k0 + kbB + i) * ldb];
        }
      }
#pragma unroll
      for (int j = 0; j < 4; ++j) {
        acc[0][j] = wmma_bf16(a0, b[j], acc[0][j]);
        acc[1][j] = wmma_bf16(a1, b[j], acc[1][j]);
      }
    }
  }
  // K tail and/or edge tiles: checked fragment loaders.
  for (; k0 < Kdim; k0 += 32) {
    FragBF a0 = load_frag_a(A, lda, row0,      k0, Mdim, Kdim);
    FragBF a1 = load_frag_a(A, lda, row0 + 16, k0, Mdim, Kdim);
#pragma unroll
    for (int j = 0; j < 4; ++j) {
      FragBF bj = BT_ROWMAJOR ? load_frag_b_nt(B, ldb, col0 + j * 16, k0, Ndim, Kdim)
                              : load_frag_b_nn(B, ldb, col0 + j * 16, k0, Ndim, Kdim);
      acc[0][j] = wmma_bf16(a0, bj, acc[0][j]);
      acc[1][j] = wmma_bf16(a1, bj, acc[1][j]);
    }
  }
  const bool chk = !full;
#pragma unroll
  for (int i = 0; i < 2; ++i)
#pragma unroll
    for (int j = 0; j < 4; ++j)
      store_tile(C, Cb, ldc, row0 + i * 16, col0 + j * 16, Mdim, Ndim, acc[i][j], bias, chk);
}

// ---------------- f32 -> bf16 convert (with optional column zero-pad) ----------------
__global__ void __launch_bounds__(256)
cvt_bf16_ker(const float* __restrict__ src, unsigned short* __restrict__ dst,
             int C, int Cp, size_t total) {
  size_t idx = (size_t)blockIdx.x * 256 + threadIdx.x;
  if (idx >= total) return;
  size_t r = idx / Cp;
  int    c = (int)(idx - r * Cp);
  dst[idx] = (c < C) ? f2bf(src[r * (size_t)C + c]) : (unsigned short)0;
}

// ---------------- transpose f32 (R x C) -> bf16 (C x ldDst), zero-pad rows >= R ----------------
__global__ void __launch_bounds__(256)
transpose_f32bf_ker(const float* __restrict__ src, int R, int C,
                    unsigned short* __restrict__ dst, int ldDst) {
  __shared__ float tile[32][33];
  const int tx = threadIdx.x & 31;
  const int ty = threadIdx.x >> 5;
  const int cb = blockIdx.x * 32;
  const int rb = blockIdx.y * 32;
#pragma unroll
  for (int i = ty; i < 32; i += 8) {
    const int r = rb + i, c = cb + tx;
    tile[i][tx] = (r < R && c < C) ? src[(size_t)r * C + c] : 0.0f;
  }
  __syncthreads();
#pragma unroll
  for (int i = ty; i < 32; i += 8) {
    const int c = cb + i, rr = rb + tx;
    if (c < C && rr < ldDst)
      dst[(size_t)c * ldDst + rr] = f2bf(tile[tx][i]);
  }
}

// ---------------- row softmax (optional mask, folded scale); bf16 or f32 output ----------------
__global__ void __launch_bounds__(256)
softmax_rows_ker(float* __restrict__ S, unsigned short* __restrict__ Ob,
                 const float* __restrict__ Msk, int cols, float scale) {
  __shared__ float red[256];
  const int r = blockIdx.x, t = threadIdx.x;
  float* row = S + (size_t)r * cols;
  const float* mrow = Msk ? (Msk + (size_t)r * cols) : nullptr;
  float mx = -3.4e38f;
  for (int c = t; c < cols; c += 256) {
    const bool on = (!mrow) || (mrow[c] > 0.5f);
    if (on) mx = fmaxf(mx, row[c] * scale);
  }
  red[t] = mx; __syncthreads();
  for (int s = 128; s > 0; s >>= 1) { if (t < s) red[t] = fmaxf(red[t], red[t + s]); __syncthreads(); }
  mx = red[0]; __syncthreads();
  float sum = 0.0f;
  for (int c = t; c < cols; c += 256) {
    const bool on = (!mrow) || (mrow[c] > 0.5f);
    const float e = on ? __expf(row[c] * scale - mx) : 0.0f;
    row[c] = e; sum += e;
  }
  red[t] = sum; __syncthreads();
  for (int s = 128; s > 0; s >>= 1) { if (t < s) red[t] += red[t + s]; __syncthreads(); }
  const float inv = 1.0f / red[0];
  if (Ob) {
    unsigned short* orow = Ob + (size_t)r * cols;
    for (int c = t; c < cols; c += 256) orow[c] = f2bf(row[c] * inv);
  } else {
    for (int c = t; c < cols; c += 256) row[c] *= inv;
  }
}

// ---------------- per-node 3-way sigmoid gates (mean over 8 heads) ----------------
__global__ void __launch_bounds__(256)
gates_ker(const float* __restrict__ Q, int dO,
          const float* __restrict__ Wgn, const float* __restrict__ bgn,
          const float* __restrict__ Wge, const float* __restrict__ bge,
          const float* __restrict__ Wgs, const float* __restrict__ bgs,
          float* __restrict__ G, int n) {
  const int i = blockIdx.x * 256 + threadIdx.x;
  if (i >= n) return;
  const float* q = Q + (size_t)i * dO;
  float an[8], ae[8], as_[8];
#pragma unroll
  for (int h = 0; h < 8; ++h) { an[h] = bgn[h]; ae[h] = bge[h]; as_[h] = bgs[h]; }
  for (int d = 0; d < dO; ++d) {
    const float qv = q[d];
    const float* wn = Wgn + (size_t)d * 8;
    const float* we = Wge + (size_t)d * 8;
    const float* ws = Wgs + (size_t)d * 8;
#pragma unroll
    for (int h = 0; h < 8; ++h) { an[h] += qv * wn[h]; ae[h] += qv * we[h]; as_[h] += qv * ws[h]; }
  }
  float gn = 0.0f, ge = 0.0f, gs = 0.0f;
#pragma unroll
  for (int h = 0; h < 8; ++h) { gn += sigmoidf_(an[h]); ge += sigmoidf_(ae[h]); gs += sigmoidf_(as_[h]); }
  G[(size_t)i * 3 + 0] = gn * 0.125f;
  G[(size_t)i * 3 + 1] = ge * 0.125f;
  G[(size_t)i * 3 + 2] = gs * 0.125f;
}

// ---------------- gated branch combine (writes f32 + bf16) ----------------
__global__ void __launch_bounds__(256)
combine_ker(const float* __restrict__ nodep, const float* __restrict__ edgep,
            const float* __restrict__ specp, const float* __restrict__ G,
            float* __restrict__ out, unsigned short* __restrict__ outb, int n, int d) {
  const size_t idx = (size_t)blockIdx.x * 256 + threadIdx.x;
  if (idx >= (size_t)n * d) return;
  const size_t i = idx / d;
  const float gn = G[i * 3 + 0], ge = G[i * 3 + 1], gsig = G[i * 3 + 2];
  const float gs = fmaxf(0.0f, 1.0f - gn - ge);
  const float tot = gn + ge + gs + 1e-8f;
  const float v = (gn * nodep[idx] + ge * edgep[idx] + gs * gsig * specp[idx]) / tot;
  out[idx] = v;
  outb[idx] = f2bf(v);
}

// ---------------- LayerNorm(Y + Res) * g + b -> f32 + bf16 (d <= 256) ----------------
__global__ void __launch_bounds__(256)
layernorm_ker(const float* __restrict__ Y, const float* __restrict__ Res,
              int d, const float* __restrict__ g, const float* __restrict__ b,
              float* __restrict__ Out, unsigned short* __restrict__ Outb) {
  __shared__ float red[256];
  const int r = blockIdx.x, t = threadIdx.x;
  const float* y  = Y   + (size_t)r * d;
  const float* rr = Res + (size_t)r * d;
  const float v = (t < d) ? (y[t] + rr[t]) : 0.0f;
  red[t] = v; __syncthreads();
  for (int s = 128; s > 0; s >>= 1) { if (t < s) red[t] += red[t + s]; __syncthreads(); }
  const float mu = red[0] / d; __syncthreads();
  const float c = (t < d) ? (v - mu) : 0.0f;
  red[t] = c * c; __syncthreads();
  for (int s = 128; s > 0; s >>= 1) { if (t < s) red[t] += red[t + s]; __syncthreads(); }
  const float rstd = rsqrtf(red[0] / d + 1e-5f);
  if (t < d) {
    const float o = c * rstd * g[t] + b[t];
    Out[(size_t)r * d + t]  = o;
    Outb[(size_t)r * d + t] = f2bf(o);
  }
}

// ---------------- column mean -> drop mask (sigmoid(ei)>0.5 <=> ei>0) ----------------
__global__ void __launch_bounds__(256)
edge_mask_ker(const float* __restrict__ P, int rows, int cols, float* __restrict__ mask) {
  const int m = blockIdx.x * 256 + threadIdx.x;
  if (m >= cols) return;
  float s = 0.0f;
  for (int i = 0; i < rows; ++i) s += P[(size_t)i * cols + m];
  mask[m] = (s / rows > 0.0f) ? 1.0f : 0.0f;
}

__global__ void __launch_bounds__(256)
hdrop_ker(const float* __restrict__ H, const float* __restrict__ mask,
          float* __restrict__ out, size_t total, int cols) {
  const size_t idx = (size_t)blockIdx.x * 256 + threadIdx.x;
  if (idx >= total) return;
  out[idx] = H[idx] * mask[idx % cols];
}

// ============================= host orchestration =============================
static void launch_gemm(bool bt, const unsigned short* A, int lda,
                        const unsigned short* B, int ldb, float* C, int ldc,
                        unsigned short* Cb, int Md, int Nd, int Kd,
                        const float* bias, hipStream_t s) {
  dim3 g((Nd + 255) / 256, (Md + 63) / 64);
  if (bt) gemm_bf16_ker<true ><<<g, 256, 0, s>>>(A, lda, B, ldb, C, ldc, Cb, Md, Nd, Kd, bias);
  else    gemm_bf16_ker<false><<<g, 256, 0, s>>>(A, lda, B, ldb, C, ldc, Cb, Md, Nd, Kd, bias);
}
static void launch_cvt(const float* src, unsigned short* dst, int R, int C, int Cp, hipStream_t s) {
  const size_t total = (size_t)R * Cp;
  cvt_bf16_ker<<<dim3((unsigned)((total + 255) / 256)), 256, 0, s>>>(src, dst, C, Cp, total);
}
static void launch_tr(const float* src, int R, int C, unsigned short* dst, int ldDst, hipStream_t s) {
  dim3 g((C + 31) / 32, (ldDst + 31) / 32);
  transpose_f32bf_ker<<<g, 256, 0, s>>>(src, R, C, dst, ldDst);
}

extern "C" void kernel_launch(void* const* d_in, const int* in_sizes, int n_in,
                              void* d_out, int out_size, void* d_ws, size_t ws_size,
                              hipStream_t stream) {
  (void)in_sizes; (void)n_in; (void)out_size;
  constexpr int Nn = 3072, Mm = 1536, FI = 1433, FIP = 1440, RANK = 32, NC = 7;
  const int dis[4]  = {512, 256, 128, 128};
  const int dos_[4] = {256, 128, 128, 128};

  const float* x     = (const float*)d_in[0];
  const float* Hin   = (const float*)d_in[1];
  const float* Ur    = (const float*)d_in[2];
  const float* cls_b = (const float*)d_in[3];
  const float* cls_w = (const float*)d_in[4];
  const float* in_b  = (const float*)d_in[5];
  const float* in_w  = (const float*)d_in[6];

  // ---- workspace bump allocator (256B aligned) ----
  char* wsp = (char*)d_ws;
  size_t off = 0;
  auto alloc = [&](size_t bytes) -> char* {
    char* p = wsp + off;
    off = (off + bytes + 255) & ~(size_t)255;
    return p;
  };
  unsigned short* xbf     = (unsigned short*)alloc((size_t)Nn * FIP * 2);
  unsigned short* Hbf     = (unsigned short*)alloc((size_t)Nn * Mm  * 2);
  unsigned short* Htbf    = (unsigned short*)alloc((size_t)Mm * Nn  * 2);
  unsigned short* Urbf    = (unsigned short*)alloc((size_t)Nn * RANK * 2);
  float*          adjf    = (float*)alloc((size_t)Nn * Nn * 4);
  float*          Sf      = (float*)alloc((size_t)Nn * Nn * 4);     // reused as P (Nn x Mm)
  unsigned short* Sbf     = (unsigned short*)alloc((size_t)Nn * Nn * 2);
  float*          h_buf   = (float*)alloc((size_t)Nn * 512 * 4);
  float*          hn_buf  = (float*)alloc((size_t)Nn * 512 * 4);
  unsigned short* hbf     = (unsigned short*)alloc((size_t)Nn * 512 * 2);
  float*          Qf      = (float*)alloc((size_t)Nn * 256 * 4);
  float*          Kf      = (float*)alloc((size_t)Nn * 256 * 4);
  float*          Vf      = (float*)alloc((size_t)Nn * 256 * 4);
  unsigned short* Qbf     = (unsigned short*)alloc((size_t)Nn * 256 * 2);
  unsigned short* Kbf     = (unsigned short*)alloc((size_t)Nn * 256 * 2);
  unsigned short* Vbf     = (unsigned short*)alloc((size_t)Nn * 256 * 2);
  float*          onode   = (float*)alloc((size_t)Nn * 256 * 4);
  float*          oedge   = (float*)alloc((size_t)Nn * 256 * 4);
  float*          specf   = (float*)alloc((size_t)Nn * 256 * 4);
  float*          comb    = (float*)alloc((size_t)Nn * 256 * 4);
  float*          proj    = (float*)alloc((size_t)Nn * 256 * 4);
  unsigned short* combbf  = (unsigned short*)alloc((size_t)Nn * 256 * 2);
  float*          he      = (float*)alloc((size_t)Mm * 512 * 4);
  unsigned short* hebf    = (unsigned short*)alloc((size_t)Mm * 512 * 2);
  float*          Qe      = (float*)alloc((size_t)Mm * 256 * 4);
  float*          Ke      = (float*)alloc((size_t)Mm * 256 * 4);
  float*          Ve      = (float*)alloc((size_t)Mm * 256 * 4);
  unsigned short* Qebf    = (unsigned short*)alloc((size_t)Mm * 256 * 2);
  unsigned short* Kebf    = (unsigned short*)alloc((size_t)Mm * 256 * 2);
  unsigned short* Vebf    = (unsigned short*)alloc((size_t)Mm * 256 * 2);
  float*          Se      = (float*)alloc((size_t)Mm * Mm * 4);
  unsigned short* Sebf    = (unsigned short*)alloc((size_t)Mm * Mm * 2);
  float*          heout   = (float*)alloc((size_t)Mm * 256 * 4);
  unsigned short* heoutbf = (unsigned short*)alloc((size_t)Mm * 256 * 2);
  float*          gatesb  = (float*)alloc((size_t)Nn * 3 * 4);
  unsigned short* wt      = (unsigned short*)alloc((size_t)512 * FIP * 2);
  if (off > ws_size) return;                         // workspace too small: bail deterministically

  float* logits = (float*)d_out;
  float* Hd     = logits + (size_t)Nn * NC;
  float* maskp  = Hd + (size_t)Nn * Mm;

  // ---- one-time conversions & adjacency ----
  launch_cvt(x,   xbf,  Nn, FI,  FIP, stream);
  launch_cvt(Hin, Hbf,  Nn, Mm,  Mm,  stream);
  launch_tr (Hin, Nn, Mm, Htbf, Nn, stream);         // H^T as (M x N) bf16
  launch_cvt(Ur,  Urbf, Nn, RANK, RANK, stream);
  launch_gemm(true, Hbf, Mm, Hbf, Mm, adjf, Nn, nullptr, Nn, Nn, Mm, nullptr, stream);  // adj = H H^T

  // ---- input projection: h = x @ in_w + in_b (f32 + bf16) ----
  launch_tr(in_w, FI, 512, wt, FIP, stream);
  launch_gemm(true, xbf, FIP, wt, FIP, h_buf, 512, hbf, Nn, 512, FIP, in_b, stream);

  float* hcur  = h_buf;
  float* hnext = hn_buf;
  for (int l = 0; l < 4; ++l) {
    const int base = 7 + l * 14;
    const float* Wge   = (const float*)d_in[base + 0];
    const float* Wgn   = (const float*)d_in[base + 1];
    const float* Wgs   = (const float*)d_in[base + 2];
    const float* Wk    = (const float*)d_in[base + 3];
    const float* Wout  = (const float*)d_in[base + 4];
    const float* Wq    = (const float*)d_in[base + 5];
    const float* Wspec = (const float*)d_in[base + 6];
    const float* Wv    = (const float*)d_in[base + 7];
    const float* bge   = (const float*)d_in[base + 8];
    const float* bgn   = (const float*)d_in[base + 9];
    const float* bgs   = (const float*)d_in[base + 10];
    const float* bout  = (const float*)d_in[base + 11];
    const float* ln_b  = (const float*)d_in[base + 12];
    const float* ln_g  = (const float*)d_in[base + 13];
    const int di = dis[l], dO = dos_[l];
    const int hd = dO / 8;
    const float scl = 1.0f / sqrtf((float)hd);

    // Q/K/V projections (f32 + bf16 fused)
    launch_tr(Wq, di, dO, wt, di, stream);
    launch_gemm(true, hbf, di, wt, di, Qf, dO, Qbf, Nn, dO, di, nullptr, stream);
    launch_tr(Wk, di, dO, wt, di, stream);
    launch_gemm(true, hbf, di, wt, di, Kf, dO, Kbf, Nn, dO, di, nullptr, stream);
    launch_tr(Wv, di, dO, wt, di, stream);
    launch_gemm(true, hbf, di, wt, di, Vf, dO, Vbf, Nn, dO, di, nullptr, stream);

    // node-level masked attention, head by head (softmax emits bf16 directly)
    for (int hh = 0; hh < 8; ++hh) {
      launch_gemm(true, Qbf + hh * hd, dO, Kbf + hh * hd, dO, Sf, Nn, nullptr, Nn, Nn, hd, nullptr, stream);
      softmax_rows_ker<<<Nn, 256, 0, stream>>>(Sf, Sbf, adjf, Nn, scl);
      launch_gemm(false, Sbf, Nn, Vbf + hh * hd, dO, onode + hh * hd, dO, nullptr, Nn, hd, Nn, nullptr, stream);
    }

    // hyperedge branch: h_e = H^T @ h, dense M x M attention
    launch_gemm(false, Htbf, Nn, hbf, di, he, di, hebf, Mm, di, Nn, nullptr, stream);
    launch_tr(Wq, di, dO, wt, di, stream);
    launch_gemm(true, hebf, di, wt, di, Qe, dO, Qebf, Mm, dO, di, nullptr, stream);
    launch_tr(Wk, di, dO, wt, di, stream);
    launch_gemm(true, hebf, di, wt, di, Ke, dO, Kebf, Mm, dO, di, nullptr, stream);
    launch_tr(Wv, di, dO, wt, di, stream);
    launch_gemm(true, hebf, di, wt, di, Ve, dO, Vebf, Mm, dO, di, nullptr, stream);
    for (int hh = 0; hh < 8; ++hh) {
      launch_gemm(true, Qebf + hh * hd, dO, Kebf + hh * hd, dO, Se, Mm, nullptr, Mm, Mm, hd, nullptr, stream);
      softmax_rows_ker<<<Mm, 256, 0, stream>>>(Se, Sebf, nullptr, Mm, scl);
      launch_gemm(false, Sebf, Mm, Vebf + hh * hd, dO, heout + hh * hd, dO, heoutbf + hh * hd,
                  Mm, hd, Mm, nullptr, stream);
    }
    launch_gemm(false, Hbf, Mm, heoutbf, dO, oedge, dO, nullptr, Nn, dO, Mm, nullptr, stream); // H @ h_e_out

    // spectral branch: spec_feat = U_r @ Wspec
    launch_tr(Wspec, RANK, dO, wt, RANK, stream);
    launch_gemm(true, Urbf, RANK, wt, RANK, specf, dO, nullptr, Nn, dO, RANK, nullptr, stream);

    // gates + combine + output projection + LN(residual); LN emits next h in f32+bf16
    gates_ker<<<(Nn + 255) / 256, 256, 0, stream>>>(Qf, dO, Wgn, bgn, Wge, bge, Wgs, bgs, gatesb, Nn);
    combine_ker<<<(unsigned)(((size_t)Nn * dO + 255) / 256), 256, 0, stream>>>(
        onode, oedge, specf, gatesb, comb, combbf, Nn, dO);
    launch_tr(Wout, dO, dO, wt, dO, stream);
    launch_gemm(true, combbf, dO, wt, dO, proj, dO, nullptr, Nn, dO, dO, bout, stream);
    layernorm_ker<<<Nn, 256, 0, stream>>>(proj, (di == dO) ? hcur : proj, dO, ln_g, ln_b, hnext, hbf);
    float* tmp = hcur; hcur = hnext; hnext = tmp;
  }

  // ---- AKED tail: attn_map = softmax(h @ h_e_out^T) (f32 in place) ----
  float* P = Sf;                                     // reuse score buffer (Nn x Mm)
  launch_gemm(true, hbf, 128, heoutbf, 128, P, Mm, nullptr, Nn, Mm, 128, nullptr, stream);
  softmax_rows_ker<<<Nn, 256, 0, stream>>>(P, nullptr, nullptr, Mm, 1.0f);
  edge_mask_ker<<<(Mm + 255) / 256, 256, 0, stream>>>(P, Nn, Mm, maskp);
  hdrop_ker<<<(unsigned)(((size_t)Nn * Mm + 255) / 256), 256, 0, stream>>>(
      Hin, maskp, Hd, (size_t)Nn * Mm, Mm);

  // ---- classifier: logits = h @ cls_w + cls_b ----
  launch_tr(cls_w, 128, NC, wt, 128, stream);
  launch_gemm(true, hbf, 128, wt, 128, logits, NC, nullptr, Nn, NC, 128, cls_b, stream);
}
